// FC_Attention_69466801045688
// MI455X (gfx1250) — matrix operand order repaired and checked by reading, not compiled, not run on hardware
//
#include <hip/hip_runtime.h>
#include <math.h>

// ---------------------------------------------------------------------------
// FC_Attention for MI455X (gfx1250, wave32, WMMA). Memory-bound problem:
// ~54 GFLOP vs ~0.4 GB HBM traffic @ 23.3 TB/s. All big GEMMs run on
// v_wmma_f32_16x16x32_bf16 with a bf16 hi/lo x3 split (fp32-class accuracy).
//
// Traffic-minimal pipeline:
//   1. colmean        : xmean[b,c]  = mean_n x[b,c,n]
//   2. bias_ln        : biasln = LN(bias_w @ xmean + bias_b)
//   3. obias          : obias = uni_b + uni_w @ biasln
//   4. gemm mode Q    : Qraw = q_w@x + q_b; epilogue emits PARTIAL per-pixel
//                       column softmax stats (max, sum exp) per 64-row tile.
//   5. gemm mode K    : k_w@x + k_b computed but NEVER stored: epilogue emits
//                       partial per-row stats + the 4 center columns only.
//   6. qstats_combine : merge 8 partials -> per-pixel (max, 1/sumexp)
//   7. kstats_combine : merge 16 partials -> per-row (max, sumexp)
//   8. k_denom        : per-batch global softmax denominator
//   9. kvp_m2         : KVp from center pixels (V computed on the fly),
//                       folded with uni_w into per-batch M2 (256x512)
//  10. gemm mode F    : out = M2[b] @ softmax(Qraw) + obias (softmax fused
//                       into the B-tile load).
// ---------------------------------------------------------------------------

#define B_    64
#define DM    256
#define N_    1024
#define DQ    512
#define DV    256
#define NH_   8
#define DQH   64
#define DVH   32
#define LN_EPS 1e-5f

typedef __attribute__((ext_vector_type(16))) __bf16 v16bf;
typedef __attribute__((ext_vector_type(8)))  float  v8f;

__device__ __forceinline__ unsigned short f2bf_bits(float f) {
    unsigned u = __builtin_bit_cast(unsigned, f);
    u += 0x7FFFu + ((u >> 16) & 1u);            // round-to-nearest-even
    return (unsigned short)(u >> 16);
}
__device__ __forceinline__ float bfbits2f(unsigned short s) {
    unsigned u = ((unsigned)s) << 16;
    return __builtin_bit_cast(float, u);
}

union FragBF { unsigned u[8]; v16bf v; };

// merge two (max, sumexp) pairs
__device__ __forceinline__ void mergeMS(float& m, float& s, float om, float os) {
    float M = fmaxf(m, om);
    s = s * __expf(m - M) + os * __expf(om - M);
    m = M;
}

// ---------------------------------------------------------------------------
// Batched GEMM: (A[b] MxKd) @ (B[b] Kd x 1024) + rowBias.
//   mode 0 (Q): store C; emit partial column stats colPart[b][M/64][n]
//   mode 1 (K): store NOTHING except center cols + partial row stats
//   mode 2 (F): B-element -> exp(v - colMax[n]) * colInvSum[n]; store C
// ---------------------------------------------------------------------------
__global__ __launch_bounds__(128) void wmma_gemm_kernel(
    const float* __restrict__ A,  long aStride,
    const float* __restrict__ Bm, long bStride,
    const float* __restrict__ rowBias, long biasStride,
    float* __restrict__ C, int M, int Kd,
    const float* __restrict__ colMax, const float* __restrict__ colInvSum,
    float2* __restrict__ colPart, float2* __restrict__ rowPart,
    float* __restrict__ Kcenter, int mode)
{
    __shared__ unsigned short As[2][64 * 34];   // [hi/lo][row][k], pad 34
    __shared__ unsigned short Bs[2][64 * 34];   // [hi/lo][col][k], pad 34
    __shared__ float sBias[64];
    __shared__ float sMax[64];
    __shared__ float sInv[64];
    __shared__ float2 qred[4][2][16];           // mode 0 cross-wave merge
    __shared__ float  kredM[2][64];             // mode 1 cross-wave merge
    __shared__ float  kredS[2][64];

    const int b   = blockIdx.z;
    const int m0  = blockIdx.y * 64;
    const int n0  = blockIdx.x * 64;
    const int tid = threadIdx.x;
    const int lane = tid & 31;
    const int wid  = tid >> 5;
    const int wm = (wid >> 1) * 32;             // wave 2x2 layout
    const int wn = (wid & 1) * 32;
    const int halfOff = (lane & 16) ? 8 : 0;

    const float* Ab = A  + (long)b * aStride;
    const float* Bb = Bm + (long)b * bStride;
    float*       Cb = C  + (long)b * (long)M * N_;

    if (tid < 64) {
        sBias[tid] = rowBias ? rowBias[(long)b * biasStride + m0 + tid] : 0.0f;
        if (mode == 2) {
            sMax[tid] = colMax[b * N_ + n0 + tid];
            sInv[tid] = colInvSum[b * N_ + n0 + tid];
        }
    }
    __syncthreads();

    v8f acc[2][2];
    #pragma unroll
    for (int mi = 0; mi < 2; ++mi)
        #pragma unroll
        for (int ni = 0; ni < 2; ++ni)
            acc[mi][ni] = (v8f)(0.0f);

    for (int kb = 0; kb < Kd; kb += 32) {
        // prefetch next k-tile (lowers to global_prefetch_b8)
        if (kb + 32 < Kd) {
            __builtin_prefetch(&Ab[(long)(m0 + (tid >> 3)) * Kd + kb + 32 + (tid & 7) * 4], 0, 1);
            __builtin_prefetch(&Bb[(long)(kb + 32 + (tid >> 4)) * N_ + n0 + (tid & 15) * 4], 0, 1);
        }
        // ---- A tile 64x32 -> LDS (hi/lo bf16), row-major --------------
        #pragma unroll
        for (int i = 0; i < 4; ++i) {
            int idx = tid + i * 128;
            int row = idx >> 3;
            int c4  = (idx & 7) * 4;
            float4 va = *(const float4*)&Ab[(long)(m0 + row) * Kd + kb + c4];
            float vv[4] = {va.x, va.y, va.z, va.w};
            #pragma unroll
            for (int e = 0; e < 4; ++e) {
                unsigned short hb = f2bf_bits(vv[e]);
                unsigned short lb = f2bf_bits(vv[e] - bfbits2f(hb));
                As[0][row * 34 + c4 + e] = hb;
                As[1][row * 34 + c4 + e] = lb;
            }
        }
        // ---- B tile 32x64 -> LDS transposed [col][k] ------------------
        #pragma unroll
        for (int i = 0; i < 4; ++i) {
            int idx = tid + i * 128;
            int kr  = idx >> 4;
            int nn  = (idx & 15) * 4;
            float4 vb = *(const float4*)&Bb[(long)(kb + kr) * N_ + n0 + nn];
            float vv[4] = {vb.x, vb.y, vb.z, vb.w};
            #pragma unroll
            for (int e = 0; e < 4; ++e) {
                float v = vv[e];
                if (mode == 2) v = __expf(v - sMax[nn + e]) * sInv[nn + e];
                unsigned short hb = f2bf_bits(v);
                unsigned short lb = f2bf_bits(v - bfbits2f(hb));
                Bs[0][(nn + e) * 34 + kr] = hb;
                Bs[1][(nn + e) * 34 + kr] = lb;
            }
        }
        __syncthreads();

        // ---- fragments per ISA 16-bit layouts -------------------------
        FragBF aHi[2], aLo[2], bHi[2], bLo[2];
        #pragma unroll
        for (int mi = 0; mi < 2; ++mi) {
            const unsigned short* bh = &As[0][(wm + mi * 16 + (lane & 15)) * 34];
            const unsigned short* bl = &As[1][(wm + mi * 16 + (lane & 15)) * 34];
            #pragma unroll
            for (int i = 0; i < 8; ++i) {
                int k = ((lane & 16) ? 8 : 0) + ((i & 4) ? 16 : 0) + ((i & 3) << 1);
                aHi[mi].u[i] = *(const unsigned*)&bh[k];
                aLo[mi].u[i] = *(const unsigned*)&bl[k];
            }
        }
        #pragma unroll
        for (int ni = 0; ni < 2; ++ni) {
            const unsigned short* bh = &Bs[0][(wn + ni * 16 + (lane & 15)) * 34];
            const unsigned short* bl = &Bs[1][(wn + ni * 16 + (lane & 15)) * 34];
            #pragma unroll
            for (int j = 0; j < 8; ++j) {
                int k = ((lane & 16) ? 8 : 0) + ((j & 4) ? 16 : 0) + ((j & 3) << 1);
                bHi[ni].u[j] = *(const unsigned*)&bh[k];
                bLo[ni].u[j] = *(const unsigned*)&bl[k];
            }
        }
        // ---- bf16x3: lo*hi + hi*lo + hi*hi ----------------------------
        #pragma unroll
        for (int mi = 0; mi < 2; ++mi)
            #pragma unroll
            for (int ni = 0; ni < 2; ++ni) {
                acc[mi][ni] = __builtin_amdgcn_wmma_f32_16x16x32_bf16(
                    false, aLo[mi].v, false, bHi[ni].v, (short)0, acc[mi][ni], false, false);
                acc[mi][ni] = __builtin_amdgcn_wmma_f32_16x16x32_bf16(
                    false, aHi[mi].v, false, bLo[ni].v, (short)0, acc[mi][ni], false, false);
                acc[mi][ni] = __builtin_amdgcn_wmma_f32_16x16x32_bf16(
                    false, aHi[mi].v, false, bHi[ni].v, (short)0, acc[mi][ni], false, false);
            }
        __syncthreads();
    }

    // ---- store C (modes 0,2). C layout: VGPR r -> M=r / M=r+8 ---------
    if (mode != 1) {
        #pragma unroll
        for (int mi = 0; mi < 2; ++mi)
            #pragma unroll
            for (int ni = 0; ni < 2; ++ni) {
                int rbase = wm + mi * 16 + halfOff;
                int col   = n0 + wn + ni * 16 + (lane & 15);
                #pragma unroll
                for (int r = 0; r < 8; ++r) {
                    int rowL = rbase + r;
                    Cb[(long)(m0 + rowL) * N_ + col] = acc[mi][ni][r] + sBias[rowL];
                }
            }
    }

    // ---- mode 0: partial per-column (max, sumexp) over this 64-row tile
    if (mode == 0) {
        #pragma unroll
        for (int ni = 0; ni < 2; ++ni) {
            float vals[16];
            float m = -3.4e38f;
            #pragma unroll
            for (int mi = 0; mi < 2; ++mi)
                #pragma unroll
                for (int r = 0; r < 8; ++r) {
                    float v = acc[mi][ni][r] + sBias[wm + mi * 16 + halfOff + r];
                    vals[mi * 8 + r] = v;
                    m = fmaxf(m, v);
                }
            float s = 0.f;
            #pragma unroll
            for (int j = 0; j < 16; ++j) s += __expf(vals[j] - m);
            // lanes l and l^16 share a column, hold complementary 8-row bands
            mergeMS(m, s, __shfl_xor(m, 16, 32), __shfl_xor(s, 16, 32));
            if (lane < 16) qred[wid][ni][lane] = make_float2(m, s);
        }
        __syncthreads();
        if (tid < 64) {     // combine wm=0 wave with wm=32 wave per column
            int c = tid, wnG = c >> 5, ni = (c >> 4) & 1, l15 = c & 15;
            float2 a = qred[wnG][ni][l15];
            float2 d = qred[wnG + 2][ni][l15];
            float m = a.x, s = a.y;
            mergeMS(m, s, d.x, d.y);
            colPart[((long)b * (M >> 6) + (m0 >> 6)) * N_ + n0 + c] = make_float2(m, s);
        }
    }

    // ---- mode 1: center columns + partial per-row (max, sumexp) -------
    if (mode == 1) {
        #pragma unroll
        for (int ni = 0; ni < 2; ++ni) {
            int colG = n0 + wn + ni * 16 + (lane & 15);
            int p = (colG == 495) ? 0 : (colG == 496) ? 1
                  : (colG == 527) ? 2 : (colG == 528) ? 3 : -1;
            if (p >= 0) {
                #pragma unroll
                for (int mi = 0; mi < 2; ++mi)
                    #pragma unroll
                    for (int r = 0; r < 8; ++r) {
                        int rowL = wm + mi * 16 + halfOff + r;
                        Kcenter[((long)b * DQ + m0 + rowL) * 4 + p] =
                            acc[mi][ni][r] + sBias[rowL];
                    }
            }
        }
        #pragma unroll
        for (int mi = 0; mi < 2; ++mi)
            #pragma unroll
            for (int r = 0; r < 8; ++r) {
                int rowL = wm + mi * 16 + halfOff + r;
                float bias = sBias[rowL];
                float v0 = acc[mi][0][r] + bias;
                float v1 = acc[mi][1][r] + bias;
                float m = fmaxf(v0, v1);
                float s = __expf(v0 - m) + __expf(v1 - m);
                // reduce across 16-lane column group (stay within half)
                #pragma unroll
                for (int off = 1; off <= 8; off <<= 1)
                    mergeMS(m, s, __shfl_xor(m, off, 32), __shfl_xor(s, off, 32));
                if ((lane & 15) == 0) { kredM[wid & 1][rowL] = m; kredS[wid & 1][rowL] = s; }
            }
        __syncthreads();
        if (tid < 64) {     // combine wn=0 wave with wn=32 wave per row
            float m = kredM[0][tid], s = kredS[0][tid];
            mergeMS(m, s, kredM[1][tid], kredS[1][tid]);
            rowPart[((long)b * DQ + m0 + tid) * (N_ >> 6) + (n0 >> 6)] = make_float2(m, s);
        }
    }
}

// ---------------------------------------------------------------------------
__global__ __launch_bounds__(256) void colmean_kernel(
    const float* __restrict__ x, float* __restrict__ xmean)
{
    const int row  = blockIdx.x * 8 + (threadIdx.x >> 5);
    const int lane = threadIdx.x & 31;
    const float* xr = x + (long)row * N_;
    float s = 0.f;
    #pragma unroll
    for (int j = 0; j < 32; ++j) s += xr[j * 32 + lane];
    #pragma unroll
    for (int off = 16; off >= 1; off >>= 1) s += __shfl_xor(s, off, 32);
    if (lane == 0) xmean[row] = s * (1.0f / N_);
}

__global__ __launch_bounds__(256) void bias_ln_kernel(
    const float* __restrict__ bias_w, const float* __restrict__ bias_b,
    const float* __restrict__ ln_g,   const float* __restrict__ ln_b,
    const float* __restrict__ xmean,  float* __restrict__ biasln)
{
    __shared__ float xm[DM];
    __shared__ float red[256];
    const int b = blockIdx.x, t = threadIdx.x;
    xm[t] = xmean[b * DM + t];
    __syncthreads();
    float bf = bias_b[t];
    for (int c = 0; c < DM; ++c) bf += bias_w[t * DM + c] * xm[c];
    red[t] = bf; __syncthreads();
    for (int s = 128; s >= 1; s >>= 1) { if (t < s) red[t] += red[t + s]; __syncthreads(); }
    float mu = red[0] * (1.0f / DV);
    __syncthreads();
    red[t] = bf * bf; __syncthreads();
    for (int s = 128; s >= 1; s >>= 1) { if (t < s) red[t] += red[t + s]; __syncthreads(); }
    float var = red[0] * (1.0f / DV) - mu * mu;
    biasln[b * DV + t] = (bf - mu) * rsqrtf(var + LN_EPS) * ln_g[t] + ln_b[t];
}

__global__ __launch_bounds__(256) void obias_kernel(
    const float* __restrict__ uni_w, const float* __restrict__ uni_b,
    const float* __restrict__ biasln, float* __restrict__ ob)
{
    __shared__ float bl[DV];
    const int b = blockIdx.x, t = threadIdx.x;
    bl[t] = biasln[b * DV + t];
    __syncthreads();
    float s = uni_b[t];
    for (int c = 0; c < DV; ++c) s += uni_w[t * DV + c] * bl[c];
    ob[b * DM + t] = s;
}

// merge the 8 per-tile column partials -> per-pixel (max, 1/sumexp)
__global__ __launch_bounds__(256) void qstats_combine_kernel(
    const float2* __restrict__ part, float* __restrict__ qMax, float* __restrict__ qInv)
{
    int idx = blockIdx.x * 256 + threadIdx.x;      // b*N_ + n
    int b = idx >> 10, n = idx & (N_ - 1);
    float2 pr[8];
    float M = -3.4e38f;
    #pragma unroll
    for (int i = 0; i < 8; ++i) {
        pr[i] = part[((long)b * 8 + i) * N_ + n];
        M = fmaxf(M, pr[i].x);
    }
    float S = 0.f;
    #pragma unroll
    for (int i = 0; i < 8; ++i) S += pr[i].y * __expf(pr[i].x - M);
    qMax[idx] = M;
    qInv[idx] = 1.0f / S;
}

// merge the 16 per-tile row partials -> per-row (max, sumexp)
__global__ __launch_bounds__(256) void kstats_combine_kernel(
    const float2* __restrict__ part, float* __restrict__ rM, float* __restrict__ rE)
{
    int row = blockIdx.x * 256 + threadIdx.x;      // b*DQ + c
    float2 pr[16];
    float M = -3.4e38f;
    #pragma unroll
    for (int i = 0; i < 16; ++i) {
        pr[i] = part[(long)row * 16 + i];
        M = fmaxf(M, pr[i].x);
    }
    float S = 0.f;
    #pragma unroll
    for (int i = 0; i < 16; ++i) S += pr[i].y * __expf(pr[i].x - M);
    rM[row] = M;
    rE[row] = S;
}

__global__ __launch_bounds__(256) void k_denom_kernel(
    const float* __restrict__ rExp, float* __restrict__ dInv)
{
    __shared__ float red[256];
    const int b = blockIdx.x, t = threadIdx.x;
    red[t] = rExp[b * DQ + t] + rExp[b * DQ + 256 + t];
    __syncthreads();
    for (int s = 128; s >= 1; s >>= 1) { if (t < s) red[t] += red[t + s]; __syncthreads(); }
    if (t == 0) dInv[b] = 1.0f / red[0];
}

// ---------------------------------------------------------------------------
// Per (b,h): KVp[q,v] = 0.25 * sum_p softmaxK[q,p] * V[v,p] at the 4 center
// pixels (V computed on the fly), folded with uni_w into M2[b,o,h*64+q].
// ---------------------------------------------------------------------------
__global__ __launch_bounds__(256) void kvp_m2_kernel(
    const float* __restrict__ x,    const float* __restrict__ v_w,
    const float* __restrict__ v_b,  const float* __restrict__ Kcenter,
    const float* __restrict__ rMax, const float* __restrict__ dInv,
    const float* __restrict__ uni_w, float* __restrict__ M2)
{
    __shared__ float xc[4][260];
    __shared__ float kc[64][5];
    __shared__ float vc[32][5];
    __shared__ float kvp[64][33];
    __shared__ float uw[256][32];
    const int h = blockIdx.x, b = blockIdx.y, t = threadIdx.x;
    const int nc[4] = {495, 496, 527, 528};   // (15,15),(15,16),(16,15),(16,16)

    #pragma unroll
    for (int i = 0; i < 4; ++i) {
        int e = t + i * 256; int c = e & 255; int p = e >> 8;
        xc[p][c] = x[((long)b * DM + c) * N_ + nc[p]];
    }
    #pragma unroll
    for (int i = 0; i < 32; ++i) {
        int e = t + i * 256; int o = e >> 5; int v = e & 31;
        uw[o][v] = uni_w[o * DV + h * DVH + v];
    }
    __syncthreads();
    if (t < 128) {
        int v = t >> 2, p = t & 3;
        float s = v_b[h * DVH + v];
        const float* wrow = &v_w[(h * DVH + v) * DM];
        for (int c = 0; c < DM; ++c) s += wrow[c] * xc[p][c];
        vc[v][p] = s;
    }
    {
        int q = t >> 2, p = t & 3;
        int ch = h * DQH + q;
        float kr = Kcenter[((long)b * DQ + ch) * 4 + p];
        kc[q][p] = __expf(kr - rMax[b * DQ + ch]) * dInv[b];
    }
    __syncthreads();
    #pragma unroll
    for (int i = 0; i < 8; ++i) {
        int e = t + i * 256; int q = e >> 5, v = e & 31;
        float s = 0.f;
        #pragma unroll
        for (int p = 0; p < 4; ++p) s += kc[q][p] * vc[v][p];
        kvp[q][v] = 0.25f * s;
    }
    __syncthreads();
    const int qq = t & 63, obk = t >> 6;
    for (int oi = 0; oi < 64; ++oi) {
        int o = obk * 64 + oi;
        float s = 0.f;
        #pragma unroll
        for (int v = 0; v < 32; ++v) s += uw[o][v] * kvp[qq][v];
        M2[((long)b * DM + o) * DQ + h * DQH + qq] = s;
    }
}

// ---------------------------------------------------------------------------
extern "C" void kernel_launch(void* const* d_in, const int* in_sizes, int n_in,
                              void* d_out, int out_size, void* d_ws, size_t ws_size,
                              hipStream_t stream)
{
    (void)in_sizes; (void)n_in; (void)out_size; (void)ws_size;
    const float* x      = (const float*)d_in[0];
    const float* q_w    = (const float*)d_in[1];
    const float* q_b    = (const float*)d_in[2];
    const float* k_w    = (const float*)d_in[3];
    const float* k_b    = (const float*)d_in[4];
    const float* v_w    = (const float*)d_in[5];
    const float* v_b    = (const float*)d_in[6];
    const float* bias_w = (const float*)d_in[7];
    const float* bias_b = (const float*)d_in[8];
    const float* ln_g   = (const float*)d_in[9];
    const float* ln_b   = (const float*)d_in[10];
    const float* uni_w  = (const float*)d_in[11];
    const float* uni_b  = (const float*)d_in[12];
    float* out = (float*)d_out;

    float* ws = (float*)d_ws;
    size_t off = 0;
    float*  Qraw  = ws + off; off += (size_t)B_ * DQ * N_;          // 128 MB
    float2* partQ = (float2*)(ws + off); off += (size_t)B_ * 8 * N_ * 2;
    float2* partK = (float2*)(ws + off); off += (size_t)B_ * DQ * 16 * 2;
    float*  Kcen  = ws + off; off += (size_t)B_ * DQ * 4;
    float*  qM    = ws + off; off += (size_t)B_ * N_;
    float*  qI    = ws + off; off += (size_t)B_ * N_;
    float*  rM    = ws + off; off += (size_t)B_ * DQ;
    float*  rE    = ws + off; off += (size_t)B_ * DQ;
    float*  dI    = ws + off; off += (size_t)B_;
    float*  xm    = ws + off; off += (size_t)B_ * DM;
    float*  bl    = ws + off; off += (size_t)B_ * DV;
    float*  ob    = ws + off; off += (size_t)B_ * DM;
    float*  M2    = ws + off; off += (size_t)B_ * DM * DQ;          // 32 MB

    colmean_kernel<<<dim3(B_ * DM / 8), 256, 0, stream>>>(x, xm);
    bias_ln_kernel<<<dim3(B_), 256, 0, stream>>>(bias_w, bias_b, ln_g, ln_b, xm, bl);
    obias_kernel<<<dim3(B_), 256, 0, stream>>>(uni_w, uni_b, bl, ob);

    // Q projection: store Qraw + partial column stats
    wmma_gemm_kernel<<<dim3(N_ / 64, DQ / 64, B_), 128, 0, stream>>>(
        q_w, 0, x, (long)DM * N_, q_b, 0, Qraw, DQ, DM,
        nullptr, nullptr, partQ, nullptr, nullptr, 0);
    // K projection: no dense store; partial row stats + center columns only
    wmma_gemm_kernel<<<dim3(N_ / 64, DQ / 64, B_), 128, 0, stream>>>(
        k_w, 0, x, (long)DM * N_, k_b, 0, nullptr, DQ, DM,
        nullptr, nullptr, nullptr, partK, Kcen, 1);

    qstats_combine_kernel<<<dim3(B_ * N_ / 256), 256, 0, stream>>>(partQ, qM, qI);
    kstats_combine_kernel<<<dim3(B_ * DQ / 256), 256, 0, stream>>>(partK, rM, rE);
    k_denom_kernel<<<dim3(B_), 256, 0, stream>>>(rE, dI);
    kvp_m2_kernel<<<dim3(NH_, B_), 256, 0, stream>>>(x, v_w, v_b, Kcen, rM, dI, uni_w, M2);

    // out = M2[b] @ softmax(Qraw[b]) + obias[b]  (softmax fused in B-load)
    wmma_gemm_kernel<<<dim3(N_ / 64, DM / 64, B_), 128, 0, stream>>>(
        M2, (long)DM * DQ, Qraw, (long)DQ * N_, ob, DM, out, DM, DQ,
        qM, qI, nullptr, nullptr, nullptr, 2);
}
